// MultiHeadAttention_3444563771659
// MI455X (gfx1250) — matrix operand order repaired and checked by reading, not compiled, not run on hardware
//
#include <hip/hip_runtime.h>
#include <hip/hip_bf16.h>

typedef __attribute__((ext_vector_type(16))) __bf16 v16bf;
typedef __attribute__((ext_vector_type(8)))  float  v8f;
typedef __attribute__((ext_vector_type(8)))  unsigned int v8u;

#define EMBED 1024
#define TOKENS 16384   // N*L = 8*2048
#define SEQ 2048
#define HEADS 16
#define HDIM 64

__device__ __forceinline__ unsigned short f2bf(float f) {
  unsigned u = __builtin_bit_cast(unsigned, f);
  u += 0x7FFFu + ((u >> 16) & 1u);           // round-to-nearest-even
  return (unsigned short)(u >> 16);
}
__device__ __forceinline__ float bf2f(unsigned short h) {
  unsigned u = ((unsigned)h) << 16;
  return __builtin_bit_cast(float, u);
}
__device__ __forceinline__ void unpack8(uint4 u, float* f) {
  f[0] = bf2f((unsigned short)(u.x)); f[1] = bf2f((unsigned short)(u.x >> 16));
  f[2] = bf2f((unsigned short)(u.y)); f[3] = bf2f((unsigned short)(u.y >> 16));
  f[4] = bf2f((unsigned short)(u.z)); f[5] = bf2f((unsigned short)(u.z >> 16));
  f[6] = bf2f((unsigned short)(u.w)); f[7] = bf2f((unsigned short)(u.w >> 16));
}

// ---- CDNA5 async DMA: global -> LDS, tracked with ASYNCcnt --------------------
__device__ __forceinline__ void async_load_b128(const void* gptr, void* lptr) {
  unsigned lds = (unsigned)(uintptr_t)lptr;
  unsigned long long ga = (unsigned long long)(uintptr_t)gptr;
  asm volatile("global_load_async_to_lds_b128 %0, %1, off"
               :: "v"(lds), "v"(ga) : "memory");
}
__device__ __forceinline__ void wait_asynccnt0() {
#if __has_builtin(__builtin_amdgcn_s_wait_asynccnt)
  __builtin_amdgcn_s_wait_asynccnt(0);
#else
  asm volatile("s_wait_asynccnt 0x0" ::: "memory");
#endif
}

// ---------------- fp32 -> bf16 bulk convert (n multiple of 2048) ---------------
__global__ __launch_bounds__(256)
void cvt_f32_to_bf16(const float* __restrict__ src, unsigned short* __restrict__ dst, int n) {
  size_t i = ((size_t)blockIdx.x * 256 + threadIdx.x) * 8;
  if (i >= (size_t)n) return;
  float4 a = *(const float4*)(src + i);
  float4 b = *(const float4*)(src + i + 4);
  uint4 o;
  o.x = (unsigned)f2bf(a.x) | ((unsigned)f2bf(a.y) << 16);
  o.y = (unsigned)f2bf(a.z) | ((unsigned)f2bf(a.w) << 16);
  o.z = (unsigned)f2bf(b.x) | ((unsigned)f2bf(b.y) << 16);
  o.w = (unsigned)f2bf(b.z) | ((unsigned)f2bf(b.w) << 16);
  *(uint4*)(dst + i) = o;
}

// ---------------- W (K x N f32) -> W^T (N x K bf16) ----------------------------
__global__ __launch_bounds__(256)
void transpose_w_bf16(const float* __restrict__ W, unsigned short* __restrict__ WT) {
  __shared__ float tile[32][33];
  const int tx = threadIdx.x & 31;
  const int ty = threadIdx.x >> 5;          // 0..7
  const int kbase = blockIdx.y * 32;
  const int nbase = blockIdx.x * 32;
#pragma unroll
  for (int i = 0; i < 32; i += 8)
    tile[ty + i][tx] = W[(size_t)(kbase + ty + i) * EMBED + nbase + tx];
  __syncthreads();
#pragma unroll
  for (int i = 0; i < 32; i += 8)
    WT[(size_t)(nbase + ty + i) * EMBED + kbase + tx] = f2bf(tile[tx][ty + i]);
}

// ---------------- bf16 WMMA GEMM:  C(MxN) = A(MxK) * B(KxN),  B given as B^T (NxK)
// block = 256 threads = 8 waves; block tile 128(M) x 256(N); waves 2(M) x 4(N);
// each wave computes a 64x64 tile = 4x4 WMMA tiles (16 v_wmma per k-step, 1:1
// with ds_load_b128). K-step 32; double-buffered LDS filled by
// GLOBAL_LOAD_ASYNC_TO_LDS_B128 (ASYNCcnt); compute overlaps the prefetch.
__global__ __launch_bounds__(256)
void gemm_bf16_wmma(const unsigned short* __restrict__ A,
                    const unsigned short* __restrict__ BT,
                    void* __restrict__ Cout,
                    const float* __restrict__ bias,
                    int M, int N, int K, int out_f32_bias)
{
  __shared__ unsigned short As[2][128 * 32];   // 2 x 8 KB
  __shared__ unsigned short Bs[2][256 * 32];   // 2 x 16 KB

  const int tid   = threadIdx.x;
  const int lane  = tid & 31;
  const int wave  = tid >> 5;
  const int waveM = wave >> 2;              // 0..1  -> 64-row span
  const int waveN = wave & 3;               // 0..3  -> 64-col span
  const int lrow  = lane & 15;
  const int lhi   = lane >> 4;

  const int bm = blockIdx.x * 128;
  const int bn = blockIdx.y * 256;

  v8f acc[4][4];
  const v8f vzero = {0.f, 0.f, 0.f, 0.f, 0.f, 0.f, 0.f, 0.f};
#pragma unroll
  for (int m = 0; m < 4; ++m)
#pragma unroll
    for (int n = 0; n < 4; ++n) acc[m][n] = vzero;

  // staging: per k-step each thread DMAs 32B of A (half a row) and 64B of B (one row)
  const int arow = tid >> 1;                // 0..127
  const int aoff = (tid & 1) * 16;          // 0 or 16 elements
  const unsigned short* ag = A  + (size_t)(bm + arow) * K + aoff;
  const unsigned short* bg = BT + (size_t)(bn + tid) * K;
  unsigned short* asl[2] = { &As[0][arow * 32 + aoff], &As[1][arow * 32 + aoff] };
  unsigned short* bsl[2] = { &Bs[0][tid * 32],         &Bs[1][tid * 32] };

  const int nk = K >> 5;

  // prologue: fetch tile 0 into buffer 0
  async_load_b128(ag,      asl[0]);
  async_load_b128(ag + 8,  asl[0] + 8);
  async_load_b128(bg,      bsl[0]);
  async_load_b128(bg + 8,  bsl[0] + 8);
  async_load_b128(bg + 16, bsl[0] + 16);
  async_load_b128(bg + 24, bsl[0] + 24);
  wait_asynccnt0();
  __syncthreads();

  for (int kt = 0; kt < nk; ++kt) {
    const int cur = kt & 1;
    // prefetch next k-tile into the other buffer while we compute
    if (kt + 1 < nk) {
      const int k1 = (kt + 1) << 5;
      async_load_b128(ag + k1,      asl[cur ^ 1]);
      async_load_b128(ag + k1 + 8,  asl[cur ^ 1] + 8);
      async_load_b128(bg + k1,      bsl[cur ^ 1]);
      async_load_b128(bg + k1 + 8,  bsl[cur ^ 1] + 8);
      async_load_b128(bg + k1 + 16, bsl[cur ^ 1] + 16);
      async_load_b128(bg + k1 + 24, bsl[cur ^ 1] + 24);
    }

    const unsigned short* Ab = As[cur];
    const unsigned short* Bb = Bs[cur];

    // A fragments: lane holds row (lane&15), K = lhi*8 + [0..7] and +16..23
    v16bf afrag[4];
#pragma unroll
    for (int m = 0; m < 4; ++m) {
      const int row = waveM * 64 + m * 16 + lrow;
      uint4 lo = *(const uint4*)&Ab[row * 32 + lhi * 8];
      uint4 hi = *(const uint4*)&Ab[row * 32 + lhi * 8 + 16];
      v8u t; t[0]=lo.x; t[1]=lo.y; t[2]=lo.z; t[3]=lo.w;
             t[4]=hi.x; t[5]=hi.y; t[6]=hi.z; t[7]=hi.w;
      afrag[m] = __builtin_bit_cast(v16bf, t);
    }
    // B fragments: lane holds col (lane&15), K = lhi*8 + [0..7] and +16..23
    v16bf bfrag[4];
#pragma unroll
    for (int n = 0; n < 4; ++n) {
      const int col = waveN * 64 + n * 16 + lrow;
      uint4 lo = *(const uint4*)&Bb[col * 32 + lhi * 8];
      uint4 hi = *(const uint4*)&Bb[col * 32 + lhi * 8 + 16];
      v8u t; t[0]=lo.x; t[1]=lo.y; t[2]=lo.z; t[3]=lo.w;
             t[4]=hi.x; t[5]=hi.y; t[6]=hi.z; t[7]=hi.w;
      bfrag[n] = __builtin_bit_cast(v16bf, t);
    }
#pragma unroll
    for (int m = 0; m < 4; ++m)
#pragma unroll
      for (int n = 0; n < 4; ++n)
        acc[m][n] = __builtin_amdgcn_wmma_f32_16x16x32_bf16(
            false, afrag[m], false, bfrag[n], (short)0, acc[m][n], false, false);

    // my async DMAs into the next buffer must land before anyone reads it
    wait_asynccnt0();
    __syncthreads();
  }

  // epilogue: C layout — VGPR r: M = lhi*8 + r, N = lane&15
#pragma unroll
  for (int m = 0; m < 4; ++m) {
#pragma unroll
    for (int n = 0; n < 4; ++n) {
      const int row = bm + waveM * 64 + m * 16 + lhi * 8;
      const int col = bn + waveN * 64 + n * 16 + lrow;
      if (out_f32_bias) {
        float* C = (float*)Cout;
        const float bb = bias[col];
#pragma unroll
        for (int r = 0; r < 8; ++r)
          C[(size_t)(row + r) * N + col] = acc[m][n][r] + bb;
      } else {
        unsigned short* C = (unsigned short*)Cout;
#pragma unroll
        for (int r = 0; r < 8; ++r)
          C[(size_t)(row + r) * N + col] = f2bf(acc[m][n][r]);
      }
    }
  }
}

// ---------------- head-gram attention (the reference's einsum "bug") -----------
// For each token t=(n,l): E[i][j] = sum_d Q[t][i*64+d]*K[t][j*64+d]; softmax over j
// scaled by 1/32; O row i = sum_j A[i][j]*V[t][j*64+..]; O stored (N,H,L,D) bf16.
__global__ __launch_bounds__(128)
void attn_headgram(const unsigned short* __restrict__ Q,
                   const unsigned short* __restrict__ K,
                   const unsigned short* __restrict__ V,
                   unsigned short* __restrict__ O)
{
  __shared__ unsigned short Ks[8 * EMBED];  // 16 KB: 8 tokens x (16 heads x 64)
  __shared__ unsigned short Vs[8 * EMBED];  // 16 KB
  const int t0  = blockIdx.x * 8;
  const int tid = threadIdx.x;

  { // async-DMA K,V rows for 8 consecutive tokens (contiguous in memory)
    const uint4* ks = (const uint4*)(K + (size_t)t0 * EMBED);
    const uint4* vs = (const uint4*)(V + (size_t)t0 * EMBED);
    uint4* kd = (uint4*)Ks;
    uint4* vd = (uint4*)Vs;
    for (int i = tid; i < 8 * EMBED / 8; i += 128) {
      async_load_b128(ks + i, kd + i);
      async_load_b128(vs + i, vd + i);
    }
  }
  wait_asynccnt0();
  __syncthreads();

  const int tok = tid >> 4;                 // 0..7
  const int row = tid & 15;                 // query head i
  const int t   = t0 + tok;

  // load own Q row (64 bf16) into registers
  float qf[64];
  {
    const uint4* qs = (const uint4*)(Q + (size_t)t * EMBED + row * HDIM);
#pragma unroll
    for (int c = 0; c < 8; ++c) { uint4 u = qs[c]; unpack8(u, &qf[c * 8]); }
  }

  float e[16];
  float mx = -1e30f;
#pragma unroll
  for (int j = 0; j < 16; ++j) {
    const uint4* kr = (const uint4*)&Ks[tok * EMBED + j * HDIM];
    float s = 0.f;
#pragma unroll
    for (int c = 0; c < 8; ++c) {
      uint4 u = kr[c];
      float f[8]; unpack8(u, f);
#pragma unroll
      for (int d = 0; d < 8; ++d) s += qf[c * 8 + d] * f[d];
    }
    e[j] = s * 0.03125f;                    // 1/sqrt(1024)
    mx = fmaxf(mx, e[j]);
  }
  float sum = 0.f;
#pragma unroll
  for (int j = 0; j < 16; ++j) { e[j] = __expf(e[j] - mx); sum += e[j]; }
  const float inv = 1.f / sum;
#pragma unroll
  for (int j = 0; j < 16; ++j) e[j] *= inv;

  // O row -> (N,H,L,D) layout so the final GEMM reads the scrambled view contiguously
  const int n = t >> 11;                    // /2048
  const int l = t & 2047;
  unsigned short* orow = O + (((size_t)(n * HEADS + row) * SEQ + l) << 6);
#pragma unroll
  for (int dc = 0; dc < HDIM; dc += 16) {
    float o[16];
#pragma unroll
    for (int d = 0; d < 16; ++d) o[d] = 0.f;
#pragma unroll
    for (int j = 0; j < 16; ++j) {
      const float a = e[j];
      const uint4* vr = (const uint4*)&Vs[tok * EMBED + j * HDIM + dc];
      uint4 u0 = vr[0], u1 = vr[1];
      float f[16]; unpack8(u0, f); unpack8(u1, &f[8]);
#pragma unroll
      for (int d = 0; d < 16; ++d) o[d] += a * f[d];
    }
    uint4 w0, w1;
    w0.x = (unsigned)f2bf(o[0])  | ((unsigned)f2bf(o[1])  << 16);
    w0.y = (unsigned)f2bf(o[2])  | ((unsigned)f2bf(o[3])  << 16);
    w0.z = (unsigned)f2bf(o[4])  | ((unsigned)f2bf(o[5])  << 16);
    w0.w = (unsigned)f2bf(o[6])  | ((unsigned)f2bf(o[7])  << 16);
    w1.x = (unsigned)f2bf(o[8])  | ((unsigned)f2bf(o[9])  << 16);
    w1.y = (unsigned)f2bf(o[10]) | ((unsigned)f2bf(o[11]) << 16);
    w1.z = (unsigned)f2bf(o[12]) | ((unsigned)f2bf(o[13]) << 16);
    w1.w = (unsigned)f2bf(o[14]) | ((unsigned)f2bf(o[15]) << 16);
    *(uint4*)(orow + dc)     = w0;
    *(uint4*)(orow + dc + 8) = w1;
  }
}

extern "C" void kernel_launch(void* const* d_in, const int* in_sizes, int n_in,
                              void* d_out, int out_size, void* d_ws, size_t ws_size,
                              hipStream_t stream) {
  const float* x  = (const float*)d_in[0];
  const float* Wv = (const float*)d_in[1];
  const float* Wk = (const float*)d_in[2];
  const float* Wq = (const float*)d_in[3];
  const float* Wo = (const float*)d_in[4];
  const float* bo = (const float*)d_in[5];
  float* out = (float*)d_out;

  // workspace carve-up (bf16 buffers)
  char* ws = (char*)d_ws;
  const size_t ACT = (size_t)TOKENS * EMBED * 2;  // 32 MB
  const size_t WT  = (size_t)EMBED * EMBED * 2;   // 2 MB
  unsigned short* xb  = (unsigned short*)ws;              ws += ACT;
  unsigned short* WvT = (unsigned short*)ws;              ws += WT;
  unsigned short* WkT = (unsigned short*)ws;              ws += WT;
  unsigned short* WqT = (unsigned short*)ws;              ws += WT;
  unsigned short* WoT = (unsigned short*)ws;              ws += WT;
  unsigned short* Qb  = (unsigned short*)ws;              ws += ACT;
  unsigned short* Kb  = (unsigned short*)ws;              ws += ACT;
  unsigned short* Vb  = (unsigned short*)ws;              ws += ACT;
  unsigned short* Ob  = (unsigned short*)ws;              ws += ACT;

  const int nx = TOKENS * EMBED;                  // 16,777,216
  cvt_f32_to_bf16<<<nx / 2048, 256, 0, stream>>>(x, xb, nx);

  dim3 tg(32, 32);
  transpose_w_bf16<<<tg, 256, 0, stream>>>(Wv, WvT);
  transpose_w_bf16<<<tg, 256, 0, stream>>>(Wk, WkT);
  transpose_w_bf16<<<tg, 256, 0, stream>>>(Wq, WqT);
  transpose_w_bf16<<<tg, 256, 0, stream>>>(Wo, WoT);

  dim3 gg(TOKENS / 128, EMBED / 256);             // 128 x 4
  gemm_bf16_wmma<<<gg, 256, 0, stream>>>(xb, WqT, Qb, nullptr, TOKENS, EMBED, EMBED, 0);
  gemm_bf16_wmma<<<gg, 256, 0, stream>>>(xb, WkT, Kb, nullptr, TOKENS, EMBED, EMBED, 0);
  gemm_bf16_wmma<<<gg, 256, 0, stream>>>(xb, WvT, Vb, nullptr, TOKENS, EMBED, EMBED, 0);

  attn_headgram<<<TOKENS / 8, 128, 0, stream>>>(Qb, Kb, Vb, Ob);

  gemm_bf16_wmma<<<gg, 256, 0, stream>>>(Ob, WoT, out, bo, TOKENS, EMBED, EMBED, 1);
}
